// ScaledDotProductAttention_46471546143194
// MI455X (gfx1250) — matrix operand order repaired and checked by reading, not compiled, not run on hardware
//
#include <hip/hip_runtime.h>
#include <hip/hip_bf16.h>
#include <math.h>

#define BB 16
#define QQ 1024
#define KK 2048
#define KD 128
#define VD 128

typedef __attribute__((ext_vector_type(16))) __bf16 v16bf;
typedef __attribute__((ext_vector_type(8)))  float  v8f;

union BO { uint4 u[2]; v16bf v; };

__device__ __forceinline__ unsigned short f2bf(float f) {
    union { float f; unsigned u; } c; c.f = f;
    unsigned u = c.u;
    // round-to-nearest-even float32 -> bf16
    return (unsigned short)((u + 0x7FFFu + ((u >> 16) & 1u)) >> 16);
}

__device__ __forceinline__ unsigned pack_bf16(float lo, float hi) {
#if __has_builtin(__builtin_amdgcn_cvt_pk_bf16_f32)
    typedef __attribute__((ext_vector_type(2))) __bf16 v2bf;
    v2bf r = __builtin_amdgcn_cvt_pk_bf16_f32(lo, hi);
    return __builtin_bit_cast(unsigned, r);
#else
    return (unsigned)f2bf(lo) | ((unsigned)f2bf(hi) << 16);
#endif
}

// ---------------- preprocessing: fp32 -> bf16 (optionally scaled) ----------------
__global__ void cvt_kernel(const float* __restrict__ in, unsigned short* __restrict__ out,
                           int n4, float scale) {
    int i = blockIdx.x * blockDim.x + threadIdx.x;
    if (i >= n4) return;
    float4 f = ((const float4*)in)[i];
    ushort4 o;
    o.x = f2bf(f.x * scale);
    o.y = f2bf(f.y * scale);
    o.z = f2bf(f.z * scale);
    o.w = f2bf(f.w * scale);
    ((ushort4*)out)[i] = o;
}

// ---------------- preprocessing: V [B,K,VD] fp32 -> V^T [B,VD,K] bf16 ----------------
__global__ void vtrans_kernel(const float* __restrict__ vals, unsigned short* __restrict__ vt) {
    __shared__ float tile[32][33];
    int b  = blockIdx.z;
    int k0 = blockIdx.x * 32;
    int v0 = blockIdx.y * 32;
    int tx = threadIdx.x;       // 0..31
    int ty = threadIdx.y;       // 0..7
    const float* src = vals + ((size_t)b * KK + k0) * VD + v0;
    #pragma unroll
    for (int i = 0; i < 32; i += 8)
        tile[ty + i][tx] = src[(size_t)(ty + i) * VD + tx];   // tile[k][vd]
    __syncthreads();
    unsigned short* dst = vt + ((size_t)b * VD + v0) * KK + k0;
    #pragma unroll
    for (int i = 0; i < 32; i += 8)
        dst[(size_t)(ty + i) * KK + tx] = f2bf(tile[tx][ty + i]); // vt[vd][k]
}

// ---------------- one 32-key chunk of online-softmax attention ----------------
// MASK=false: all 32 keys known valid (hot loop, no compares/cndmasks).
// MASK=true : per-key validity against nk (final partial chunk only).
template<bool MASK>
__device__ __forceinline__ void attn_chunk(
    int k0, int nk, int ln, int lh,
    const unsigned short* __restrict__ kbb,
    const unsigned short* __restrict__ vtb,
    const BO* __restrict__ qop,
    v8f* __restrict__ acc,
    float& mrow, float& lsum)
{
    // ---- S^T = K * Q^T : two 16(key)x16(query) tiles, reduce kd in 4 WMMAs each
    // A layout (16x32 bf16): lane ln = key row; element e -> kd = c*32 + (e>=8?16:0) + 8*lh + (e&7).
    v8f s0 = (v8f){0.f,0.f,0.f,0.f,0.f,0.f,0.f,0.f};
    v8f s1 = s0;
    const unsigned short* krow0 = kbb + (size_t)(k0 + ln) * KD + lh * 8;
    const unsigned short* krow1 = kbb + (size_t)(k0 + 16 + ln) * KD + lh * 8;
    #pragma unroll
    for (int c = 0; c < 4; ++c) {
        BO a0, a1;
        a0.u[0] = *(const uint4*)(krow0 + c * 32);
        a0.u[1] = *(const uint4*)(krow0 + c * 32 + 16);
        a1.u[0] = *(const uint4*)(krow1 + c * 32);
        a1.u[1] = *(const uint4*)(krow1 + c * 32 + 16);
        s0 = __builtin_amdgcn_wmma_f32_16x16x32_bf16(false, a0.v, false, qop[c].v,
                                                     (short)0, s0, false, false);
        s1 = __builtin_amdgcn_wmma_f32_16x16x32_bf16(false, a1.v, false, qop[c].v,
                                                     (short)0, s1, false, false);
    }

    // ---- online softmax (C layout: lane ln = query col; reg r -> key 8*lh + r)
    float p0[8], p1[8];
    float mx = -INFINITY;
    #pragma unroll
    for (int r = 0; r < 8; ++r) {
        float v0 = s0[r], v1 = s1[r];
        if (MASK) {
            const int kb0 = k0 + 8 * lh + r;
            v0 = (kb0      < nk) ? v0 : -INFINITY;
            v1 = (kb0 + 16 < nk) ? v1 : -INFINITY;
        }
        p0[r] = v0; p1[r] = v1;
        mx = fmaxf(mx, fmaxf(v0, v1));
    }
    mx = fmaxf(mx, __shfl_xor(mx, 16, 32));      // combine key halves (same query)
    const float mnew  = fmaxf(mrow, mx);
    const float alpha = __expf(mrow - mnew);      // 0 on first chunk (exp(-inf))
    float sl = 0.0f;
    #pragma unroll
    for (int r = 0; r < 8; ++r) {
        p0[r] = __expf(p0[r] - mnew);             // masked entries -> exp(-inf)=0
        p1[r] = __expf(p1[r] - mnew);
        sl += p0[r] + p1[r];
    }
    sl += __shfl_xor(sl, 16, 32);
    lsum = fmaf(lsum, alpha, sl);
    mrow = mnew;

    // rescale existing accumulators (vector form -> packed/dual-issue f32 muls)
    #pragma unroll
    for (int t = 0; t < 8; ++t)
        acc[t] = acc[t] * alpha;

    // ---- build P^T B-operand (lane ln = query col; element e -> key 16*lh + e).
    // Pack float pairs -> bf16x2 dwords, then one half-wave swap per dword.
    unsigned d0[4], d1[4];
    #pragma unroll
    for (int i = 0; i < 4; ++i) {
        d0[i] = pack_bf16(p0[2*i], p0[2*i+1]);
        d1[i] = pack_bf16(p1[2*i], p1[2*i+1]);
    }
    union PB { unsigned d[8]; v16bf v; } pb;
    #pragma unroll
    for (int i = 0; i < 4; ++i) {
        unsigned t0 = (unsigned)__shfl_xor((int)d0[i], 16, 32);
        unsigned t1 = (unsigned)__shfl_xor((int)d1[i], 16, 32);
        pb.d[i]     = lh ? t1 : d0[i];   // keys 16*lh + {0..7}
        pb.d[i + 4] = lh ? d1[i] : t0;   // keys 16*lh + {8..15}
    }

    // ---- O^T += V^T * P^T : 8 vd-tiles, one 16x16x32 WMMA each
    const unsigned short* vrow = vtb + (size_t)ln * KK + k0 + 16 * lh;
    #pragma unroll
    for (int t = 0; t < 8; ++t) {
        BO av;
        const unsigned short* vp = vrow + (size_t)t * 16 * KK;
        av.u[0] = *(const uint4*)(vp);
        av.u[1] = *(const uint4*)(vp + 8);
        acc[t] = __builtin_amdgcn_wmma_f32_16x16x32_bf16(false, av.v, false, pb.v,
                                                         (short)0, acc[t], false, false);
    }
}

// ---------------- fused flash attention (bf16 WMMA, fp32 accumulate) ----------------
// 1 wave = 16 queries x full VD=128.  4 waves / block.  Computes S^T = K*Q^T and
// O^T = V^T * P^T so both GEMMs' operands match the CDNA5 wave32 A/B VGPR layouts
// with only b128 loads; P^T is built from the S^T accumulators with 8 half-wave
// shuffles (no LDS round trip).  Full 32-key chunks run with no masking; only the
// final partial chunk (wave-uniform branch) pays compare/cndmask cost.
__global__ __launch_bounds__(128) void attn_kernel(
    const unsigned short* __restrict__ qb,   // [B,Q,KD]  bf16, pre-scaled by 1/sqrt(KD)
    const unsigned short* __restrict__ kb,   // [B,K,KD]  bf16
    const unsigned short* __restrict__ vt,   // [B,VD,K]  bf16 (V transposed)
    const long long* __restrict__ num_keys,  // [B] int64
    float* __restrict__ out)                 // [B,Q,VD]  fp32
{
    const int wave = threadIdx.x >> 5;
    const int lane = threadIdx.x & 31;
    const int ln   = lane & 15;   // N / M index within a 16-wide tile
    const int lh   = lane >> 4;   // lane half

    const int b  = blockIdx.x >> 4;                       // 16 blocks per batch
    const int q0 = ((blockIdx.x & 15) * 4 + wave) * 16;   // 16-query tile base
    const int nk = (int)num_keys[b];

    // Preload Q^T B-operand, all 4 kd-chunks of 32.
    // B layout (32x16 bf16): lane ln = query column; element e -> kd = c*32 + (e>=8?16:0) + 8*lh + (e&7).
    const unsigned short* qrow = qb + ((size_t)b * QQ + q0 + ln) * KD;
    BO qop[4];
    #pragma unroll
    for (int c = 0; c < 4; ++c) {
        const uint4* p = (const uint4*)(qrow + c * 32 + lh * 8);
        qop[c].u[0] = p[0];
        qop[c].u[1] = p[2];
    }

    v8f acc[8];   // O^T accumulators: 8 vd-tiles of 16x16 f32
    #pragma unroll
    for (int t = 0; t < 8; ++t) acc[t] = (v8f){0.f,0.f,0.f,0.f,0.f,0.f,0.f,0.f};
    float mrow = -INFINITY;   // running row max (per query column = per lane pair)
    float lsum = 0.0f;        // running row sum

    const unsigned short* kbb = kb + (size_t)b * KK * KD;
    const unsigned short* vtb = vt + (size_t)b * VD * KK;

    const int nfull = nk >> 5;          // full 32-key chunks (no masking needed)
    for (int kt = 0; kt < nfull; ++kt)
        attn_chunk<false>(kt << 5, nk, ln, lh, kbb, vtb, qop, acc, mrow, lsum);
    if (nk & 31)                         // wave-uniform: nk identical across the wave
        attn_chunk<true>(nfull << 5, nk, ln, lh, kbb, vtb, qop, acc, mrow, lsum);

    // ---- epilogue: O^T C layout -> lane ln = query; reg r -> vd = t*16 + 8*lh + r
    const float inv = 1.0f / lsum;
    float* orow = out + ((size_t)b * QQ + q0 + ln) * VD + 8 * lh;
    #pragma unroll
    for (int t = 0; t < 8; ++t) {
        float4 o0 = make_float4(acc[t][0]*inv, acc[t][1]*inv, acc[t][2]*inv, acc[t][3]*inv);
        float4 o1 = make_float4(acc[t][4]*inv, acc[t][5]*inv, acc[t][6]*inv, acc[t][7]*inv);
        *(float4*)(orow + t * 16)     = o0;
        *(float4*)(orow + t * 16 + 4) = o1;
    }
}

extern "C" void kernel_launch(void* const* d_in, const int* in_sizes, int n_in,
                              void* d_out, int out_size, void* d_ws, size_t ws_size,
                              hipStream_t stream) {
    const float*     qrys  = (const float*)d_in[0];
    const float*     keys  = (const float*)d_in[1];
    const float*     vals  = (const float*)d_in[2];
    const long long* nkeys = (const long long*)d_in[3];   // int64 per reference
    float* out = (float*)d_out;

    // workspace layout: qb (4MB) | kb (8MB) | vt (8MB)  == 20MB bf16
    unsigned short* qb = (unsigned short*)d_ws;
    unsigned short* kb = qb + (size_t)BB * QQ * KD;
    unsigned short* vt = kb + (size_t)BB * KK * KD;

    const float scale = 0.08838834764831845f;  // 1/sqrt(128), folded into Q

    int nq4 = BB * QQ * KD / 4;
    cvt_kernel<<<(nq4 + 255) / 256, 256, 0, stream>>>(qrys, qb, nq4, scale);
    int nk4 = BB * KK * KD / 4;
    cvt_kernel<<<(nk4 + 255) / 256, 256, 0, stream>>>(keys, kb, nk4, 1.0f);
    vtrans_kernel<<<dim3(KK / 32, VD / 32, BB), dim3(32, 8), 0, stream>>>(vals, vt);

    // 1 wave per 16 queries, 4 waves per block: B*Q/64 = 256 blocks
    attn_kernel<<<dim3(BB * QQ / 64), 128, 0, stream>>>(qb, kb, vt, nkeys, out);
}
